// MlpwithSOMModule_8151847928469
// MI455X (gfx1250) — matrix operand order repaired and checked by reading, not compiled, not run on hardware
//
#include <hip/hip_runtime.h>
#include <hip/hip_bf16.h>

typedef __attribute__((ext_vector_type(16))) _Float16 v16h;
typedef __attribute__((ext_vector_type(8)))  _Float16 v8h;
typedef __attribute__((ext_vector_type(8)))  float    v8f;

#define TT   128   // T
#define DD   64    // D
#define HH   128   // hidden = 2*D
#define ROWS 256   // pair-rows per block = 2 context rows * 128 entities

union V16U { v16h v; v8h h[2]; };

// CDNA5 hardware tanh (TRANS32 pipe). v_nop after per ISA trans-op hazard rule.
__device__ __forceinline__ float fast_tanh(float x) {
    float r;
    asm volatile("v_tanh_f32 %0, %1\n\tv_nop" : "=v"(r) : "v"(x));
    return r;
}

// A fragment (16x32 f16, M x K) from row-major LDS [row][HH] halfs.
// Lanes 0-15: M=lane, halfs = K {k0+0..7, k0+16..23}
// Lanes 16-31: M=lane-16, halfs = K {k0+8..15, k0+24..31}
__device__ __forceinline__ v16h load_frag_A(const _Float16* __restrict__ smX,
                                            int lane, int row0, int k0) {
    const int hs = lane >> 4;
    const int r  = lane & 15;
    const _Float16* p = smX + (row0 + r) * HH + k0 + hs * 8;
    V16U t;
    t.h[0] = *(const v8h*)(p);        // K = k0 + hs*8 .. +7
    t.h[1] = *(const v8h*)(p + 16);   // K = k0 + 16 + hs*8 .. +7
    return t.v;
}

// B fragment (32x16 f16, K x N) from transposed weights LDS [n][HH] halfs.
// Lanes 0-15: N=lane, K={k0+0..7, k0+16..23} ; Lanes 16-31: N=lane-16, K={k0+8..15, k0+24..31}
__device__ __forceinline__ v16h load_frag_B(const _Float16* __restrict__ smW,
                                            int lane, int n0, int k0) {
    const int hs = lane >> 4;
    const int n  = lane & 15;
    const _Float16* p = smW + (n0 + n) * HH + k0 + hs * 8;
    V16U t;
    t.h[0] = *(const v8h*)(p);
    t.h[1] = *(const v8h*)(p + 16);
    return t.v;
}

// Kernel 0: convert 6x128x128 f32 weights to f16, transposed [l][n][k].
__global__ void wprep_kernel(const float* __restrict__ Ws, _Float16* __restrict__ wt) {
    int idx = blockIdx.x * blockDim.x + threadIdx.x;   // output index
    int l = idx >> 14;
    int n = (idx >> 7) & 127;
    int k = idx & 127;
    wt[idx] = (_Float16)Ws[(l << 14) + (k << 7) + n];
}

// Kernel 1: per block: group g, 2 context rows, all 128 entities.
__global__ __launch_bounds__(256) void
mlp_pairs_kernel(const float* __restrict__ input,
                 const _Float16* __restrict__ wt,
                 const float* __restrict__ bs,
                 const float* __restrict__ W_out,
                 const float* __restrict__ b_out,
                 float* __restrict__ rowmax) {
    __shared__ __align__(16) _Float16 smX[ROWS * HH];   // 64 KB activations (in-place)
    __shared__ __align__(16) _Float16 smW[HH * HH];     // 32 KB layer weights (transposed)
    __shared__ float smBias[HH];
    __shared__ float smWo[HH];
    __shared__ float smScore[ROWS];

    const int tid  = threadIdx.x;
    const int lane = tid & 31;
    const int wave = tid >> 5;
    const int g    = blockIdx.x >> 6;
    const int i0   = (blockIdx.x & 63) << 1;

    // ---- Build X0: row p = il*128 + j ; [0:64]=f16(ctx[i0+il]), [64:128]=f16(ent[j])
    {
        const int p  = tid;
        const int il = p >> 7;
        const int j  = p & 127;
        const float* ctx = input + ((size_t)g * 2 + 0) * TT * DD + (size_t)(i0 + il) * DD;
        const float* ent = input + ((size_t)g * 2 + 1) * TT * DD + (size_t)j * DD;
        _Float16* dst = smX + p * HH;
        #pragma unroll 8
        for (int d = 0; d < DD; ++d) dst[d] = (_Float16)ctx[d];
        #pragma unroll 8
        for (int d = 0; d < DD; ++d) dst[DD + d] = (_Float16)ent[d];
    }
    if (tid < HH) smWo[tid] = W_out[tid];

    const int mBase = wave * 32;   // each wave owns 32 exclusive pair-rows

    for (int l = 0; l < 6; ++l) {
        // Stage layer weights (f16, [n][k]) and bias into LDS.
        {
            const v8h* src = (const v8h*)(wt + (size_t)l * HH * HH);
            v8h* dstw = (v8h*)smW;
            #pragma unroll
            for (int idx = tid; idx < (HH * HH) / 8; idx += 256) dstw[idx] = src[idx];
        }
        if (tid < HH) smBias[tid] = bs[l * HH + tid];
        __syncthreads();

        // 32x128 output tile per wave: 2 M-tiles x 8 N-tiles of 16x16.
        v8f acc[2][8];
        #pragma unroll
        for (int mi = 0; mi < 2; ++mi)
            #pragma unroll
            for (int ni = 0; ni < 8; ++ni) acc[mi][ni] = (v8f)(0.0f);

        #pragma unroll
        for (int kk = 0; kk < 4; ++kk) {
            const int k0 = kk * 32;
            v16h a[2];
            #pragma unroll
            for (int mi = 0; mi < 2; ++mi)
                a[mi] = load_frag_A(smX, lane, mBase + mi * 16, k0);
            #pragma unroll
            for (int ni = 0; ni < 8; ++ni) {
                v16h b = load_frag_B(smW, lane, ni * 16, k0);
                #pragma unroll
                for (int mi = 0; mi < 2; ++mi)
                    acc[mi][ni] = __builtin_amdgcn_wmma_f32_16x16x32_f16(
                        false, a[mi], false, b, (short)0, acc[mi][ni], false, false);
            }
        }

        // Epilogue: bias + hardware tanh, write back in place (rows are wave-exclusive).
        {
            const int hs = lane >> 4;
            const int ln = lane & 15;
            #pragma unroll
            for (int mi = 0; mi < 2; ++mi) {
                #pragma unroll
                for (int ni = 0; ni < 8; ++ni) {
                    const int ncol = ni * 16 + ln;
                    const float bias = smBias[ncol];
                    #pragma unroll
                    for (int r = 0; r < 8; ++r) {
                        const int m = mBase + mi * 16 + hs * 8 + r;
                        smX[m * HH + ncol] = (_Float16)fast_tanh(acc[mi][ni][r] + bias);
                    }
                }
            }
        }
        __syncthreads();   // all reads of smX/smW for this layer done before next stage
    }

    // ---- Final projection: score[p] = dot(x[p], W_out) + b_out
    {
        float accv = 0.0f;
        const _Float16* xr = smX + tid * HH;
        #pragma unroll 16
        for (int k = 0; k < HH; ++k) accv = fmaf((float)xr[k], smWo[k], accv);
        smScore[tid] = accv + b_out[0];
    }
    __syncthreads();

    // ---- max over j (128 per context row), apply mask, write per-(g,i) result
    if (tid < 2) {
        const int i = i0 + tid;
        const float* sc = smScore + tid * 128;
        float m = sc[0];
        #pragma unroll 16
        for (int j = 1; j < 128; ++j) m = fmaxf(m, sc[j]);
        const float c0 = input[((size_t)g * 2 + 0) * TT * DD + (size_t)i * DD];
        rowmax[g * TT + i] = (c0 != 0.0f) ? m : 0.0f;
    }
}

// Kernel 2: out[g] = sum_i rowmax[g][i]  (deterministic serial reduce)
__global__ void reduce_kernel(const float* __restrict__ rowmax, float* __restrict__ out, int G) {
    int g = threadIdx.x;
    if (g < G) {
        float s = 0.0f;
        #pragma unroll 16
        for (int i = 0; i < TT; ++i) s += rowmax[g * TT + i];
        out[g] = s;
    }
}

extern "C" void kernel_launch(void* const* d_in, const int* in_sizes, int n_in,
                              void* d_out, int out_size, void* d_ws, size_t ws_size,
                              hipStream_t stream) {
    const float* input = (const float*)d_in[0];
    const float* Ws    = (const float*)d_in[1];
    const float* bs    = (const float*)d_in[2];
    const float* W_out = (const float*)d_in[3];
    const float* b_out = (const float*)d_in[4];
    float* out = (float*)d_out;

    const int G = in_sizes[0] / (2 * TT * DD);   // 32 for the reference shapes

    _Float16* wt = (_Float16*)d_ws;                                  // 6*128*128 f16 = 192 KB
    float* rowmax = (float*)((char*)d_ws + (size_t)6 * HH * HH * sizeof(_Float16));

    wprep_kernel<<<(6 * HH * HH) / 256, 256, 0, stream>>>(Ws, wt);
    mlp_pairs_kernel<<<G * 64, 256, 0, stream>>>(input, wt, bs, W_out, b_out, rowmax);
    reduce_kernel<<<1, 64, 0, stream>>>(rowmax, out, G);
}
